// TopKSAE_16492674416837
// MI455X (gfx1250) — matrix operand order repaired and checked by reading, not compiled, not run on hardware
//
#include <hip/hip_runtime.h>

#define D_IN   768
#define D_SAE  24576
#define BATCH  4096
#define KSEL   64

typedef __attribute__((ext_vector_type(16))) _Float16 v16h;
typedef __attribute__((ext_vector_type(8)))  _Float16 v8h;
typedef __attribute__((ext_vector_type(2)))  _Float16 v2h;
typedef __attribute__((ext_vector_type(8)))  float    v8f;
typedef __attribute__((ext_vector_type(4)))  int      v4i;

#if defined(__gfx1250__) && __has_builtin(__builtin_amdgcn_global_load_async_to_lds_b128)
#define HAVE_ASYNC_LDS 1
typedef __attribute__((address_space(1))) v4i glob_v4i;
typedef __attribute__((address_space(3))) v4i lds_v4i;
#else
#define HAVE_ASYNC_LDS 0
#endif

// ---------------------------------------------------------------------------
// Kernel 1: encoder GEMM  h = (x - b_dec) @ W_enc + b_enc   via WMMA f16->f32
//   WG: 256 threads (8 wave32).  WG tile: 64(M) x 128(N).  K step: 32.
//   Software pipeline: register-staged global loads + double-buffered LDS.
// ---------------------------------------------------------------------------
#define BM 64
#define BN 128
#define BK 32
#define LDA 56   // halves; 112B row stride: 16B-aligned b128 frags, 28-bank step
#define LDB 56

__global__ __launch_bounds__(256)
void sae_encode_gemm(const float* __restrict__ x,
                     const float* __restrict__ Wenc,
                     const float* __restrict__ benc,
                     const float* __restrict__ bdec,
                     float* __restrict__ h)
{
    __shared__ _Float16 Ash[2][BM * LDA];   // [m][k], k contiguous
    __shared__ _Float16 Bsh[2][BN * LDB];   // [n][k], k contiguous (transposed fill)

    const int tid  = threadIdx.x;
    const int lane = tid & 31;
    const int wid  = tid >> 5;

    const int nBase = blockIdx.x * BN;
    const int mBase = blockIdx.y * BM;

    const int mW = (wid & 3) * 16;   // wave M offset in WG tile
    const int nW = (wid >> 2) * 64;  // wave N offset in WG tile

    v8f acc[4];
    #pragma unroll
    for (int f = 0; f < 4; ++f)
        #pragma unroll
        for (int i = 0; i < 8; ++i) acc[f][i] = 0.0f;

    // A fill: thread -> (m = tid/4, 8 contiguous k at (tid%4)*8)
    const int am = tid >> 2;
    const int ak = (tid & 3) * 8;
    // B fill: thread -> k-pair kp (k = 2kp, 2kp+1), 4 columns at bn (+64 on it=1)
    const int kp = tid & 15;
    const int bn = (tid >> 4) * 4;

    // Fragment addressing per documented 16-bit WMMA wave32 layouts:
    //  A 16x32: lanes<16 hold k 0..7 / 16..23 ; lanes>=16 hold k 8..15 / 24..31
    //  B 32x16: lanes<16 hold k 0..7 / 16..23 ; lanes>=16 hold k 8..15 / 24..31 (col = lane%16)
    const int rA   = lane & 15;
    const int kaA  = (lane < 16) ? 0 : 8;
    const int colB = lane & 15;
    const int kbB  = (lane < 16) ? 0 : 8;

    const int NT = D_IN / BK;   // 24 K-tiles

    float4 ax0, ax1, ab0, ab1;      // staged A + b_dec
    float4 bw00, bw01, bw10, bw11;  // staged W rows: [it][k / k+1]

    auto issue_loads = [&](int t) {
        const int kb = t * BK;
        const float* xp = x + (size_t)(mBase + am) * D_IN + kb + ak;
        ax0 = *(const float4*)(xp);
        ax1 = *(const float4*)(xp + 4);
        ab0 = *(const float4*)(bdec + kb + ak);
        ab1 = *(const float4*)(bdec + kb + ak + 4);
        const float* wp0 = Wenc + (size_t)(kb + 2 * kp) * D_SAE + nBase + bn;
        bw00 = *(const float4*)(wp0);
        bw01 = *(const float4*)(wp0 + D_SAE);
        bw10 = *(const float4*)(wp0 + 64);
        bw11 = *(const float4*)(wp0 + D_SAE + 64);
        if (t + 1 < NT)  // keep the next W tile moving toward L2
            __builtin_prefetch(wp0 + (size_t)BK * D_SAE, 0, 1);
    };

    auto stage_store = [&](int buf) {
        _Float16* da = &Ash[buf][am * LDA + ak];
        da[0] = (_Float16)(ax0.x - ab0.x);
        da[1] = (_Float16)(ax0.y - ab0.y);
        da[2] = (_Float16)(ax0.z - ab0.z);
        da[3] = (_Float16)(ax0.w - ab0.w);
        da[4] = (_Float16)(ax1.x - ab1.x);
        da[5] = (_Float16)(ax1.y - ab1.y);
        da[6] = (_Float16)(ax1.z - ab1.z);
        da[7] = (_Float16)(ax1.w - ab1.w);
        v2h p;
        #define BPACK(n0, wa, wb, c) \
            p[0] = (_Float16)wa.c; p[1] = (_Float16)wb.c; \
            *(v2h*)&Bsh[buf][(n0) * LDB + 2 * kp] = p;
        BPACK(bn + 0, bw00, bw01, x) BPACK(bn + 1, bw00, bw01, y)
        BPACK(bn + 2, bw00, bw01, z) BPACK(bn + 3, bw00, bw01, w)
        BPACK(bn + 64, bw10, bw11, x) BPACK(bn + 65, bw10, bw11, y)
        BPACK(bn + 66, bw10, bw11, z) BPACK(bn + 67, bw10, bw11, w)
        #undef BPACK
    };

    issue_loads(0);
    stage_store(0);
    __syncthreads();

    int cur = 0;
    for (int t = 0; t < NT; ++t) {
        if (t + 1 < NT) issue_loads(t + 1);   // globals in flight during compute

        // ---- all fragment LDS loads up front, then WMMAs back-to-back ----
        const v8h alo = *(const v8h*)&Ash[cur][(mW + rA) * LDA + kaA];
        const v8h ahi = *(const v8h*)&Ash[cur][(mW + rA) * LDA + kaA + 16];
        v8h blo[4], bhi[4];
        #pragma unroll
        for (int f = 0; f < 4; ++f) {
            blo[f] = *(const v8h*)&Bsh[cur][(nW + f * 16 + colB) * LDB + kbB];
            bhi[f] = *(const v8h*)&Bsh[cur][(nW + f * 16 + colB) * LDB + kbB + 16];
        }
        v16h afrag;
        #pragma unroll
        for (int i = 0; i < 8; ++i) { afrag[i] = alo[i]; afrag[8 + i] = ahi[i]; }
        #pragma unroll
        for (int f = 0; f < 4; ++f) {
            v16h bfrag;
            #pragma unroll
            for (int i = 0; i < 8; ++i) { bfrag[i] = blo[f][i]; bfrag[8 + i] = bhi[f][i]; }
            acc[f] = __builtin_amdgcn_wmma_f32_16x16x32_f16(
                false, afrag, false, bfrag, (short)0, acc[f], false, false);
        }

        if (t + 1 < NT) stage_store(cur ^ 1);
        __syncthreads();
        cur ^= 1;
    }

    // ---- epilogue: C/D layout row = j + 8*(lane>=16), col = lane%16 ----
    const int rowOff = (lane < 16) ? 0 : 8;
    #pragma unroll
    for (int f = 0; f < 4; ++f) {
        const int colg = nBase + nW + f * 16 + colB;
        const float bias = benc[colg];
        float* outp = h + (size_t)(mBase + mW + rowOff) * D_SAE + colg;
        #pragma unroll
        for (int j = 0; j < 8; ++j)
            outp[(size_t)j * D_SAE] = acc[f][j] + bias;
    }
}

// ---------------------------------------------------------------------------
// Kernel 2: per-row exact top-64 (radix select on sortable keys), in-place
// rewrite of the row to relu(topk) / 0, and compacted (idx,val) pairs to ws.
// Row is pulled into LDS with async global->LDS DMA (ASYNCcnt) when available.
// ---------------------------------------------------------------------------
__global__ __launch_bounds__(256)
void sae_topk_select(float* __restrict__ h, float* __restrict__ pairs)
{
    extern __shared__ unsigned int keys[];            // D_SAE entries (96KB)
    __shared__ unsigned int hist[256];
    __shared__ unsigned int eqbase[256];
    __shared__ unsigned int sh_prefix, sh_remaining, sh_nsel;

    const int row = blockIdx.x;
    const int tid = threadIdx.x;
    float* hrow = h + (size_t)row * D_SAE;

#if HAVE_ASYNC_LDS
    // async DMA the whole row (raw f32 bits) into LDS: 16B per lane per issue
    for (int i = tid * 4; i < D_SAE; i += 1024) {
        __builtin_amdgcn_global_load_async_to_lds_b128(
            (glob_v4i*)(hrow + i),
            (lds_v4i*)(keys + i),
            0, 0);
    }
#if __has_builtin(__builtin_amdgcn_s_wait_asynccnt)
    __builtin_amdgcn_s_wait_asynccnt(0);
#else
    asm volatile("s_wait_asynccnt 0x0" ::: "memory");
#endif
    if (tid == 0) { sh_prefix = 0u; sh_remaining = KSEL; sh_nsel = 0u; }
    __syncthreads();
    // transform raw bits -> monotone-sortable keys, in place
    for (int i = tid; i < D_SAE; i += 256) {
        unsigned int b = keys[i];
        keys[i] = (b & 0x80000000u) ? ~b : (b | 0x80000000u);
    }
#else
    for (int i = tid; i < D_SAE; i += 256) {
        unsigned int b = __float_as_uint(hrow[i]);
        keys[i] = (b & 0x80000000u) ? ~b : (b | 0x80000000u);
    }
    if (tid == 0) { sh_prefix = 0u; sh_remaining = KSEL; sh_nsel = 0u; }
#endif
    __syncthreads();

    // 4-pass MSB-first radix select for the 64th-largest key
    for (int shift = 24; shift >= 0; shift -= 8) {
        hist[tid] = 0u;
        __syncthreads();
        const unsigned int pre = sh_prefix;
        const unsigned int hm  = (shift == 24) ? 0u : (0xFFFFFFFFu << (shift + 8));
        for (int i = tid; i < D_SAE; i += 256) {
            unsigned int k = keys[i];
            if ((k & hm) == pre) atomicAdd(&hist[(k >> shift) & 255u], 1u);
        }
        __syncthreads();
        if (tid == 0) {
            unsigned int rem = sh_remaining, cum = 0u;
            int bsel = 0;
            for (int b = 255; b >= 0; --b) {
                unsigned int c = hist[b];
                if (cum + c >= rem) { bsel = b; sh_remaining = rem - cum; break; }
                cum += c;
            }
            sh_prefix = pre | ((unsigned int)bsel << shift);
        }
        __syncthreads();
    }
    const unsigned int T = sh_prefix;      // exact key of 64th largest
    const unsigned int r = sh_remaining;   // how many ties (==T) to take

    // tie-break by lowest index: contiguous chunks + rank scan over equals
    const int CH = D_SAE / 256;            // 96
    const int base = tid * CH;
    unsigned int cnt = 0u;
    for (int i = 0; i < CH; ++i) cnt += (keys[base + i] == T);
    hist[tid] = cnt;
    __syncthreads();
    if (tid == 0) {
        unsigned int s = 0u;
        for (int t2 = 0; t2 < 256; ++t2) { eqbase[t2] = s; s += hist[t2]; }
    }
    __syncthreads();

    unsigned int rank = eqbase[tid];
    for (int i = 0; i < CH; ++i) {
        const int idx = base + i;
        const unsigned int k = keys[idx];
        bool sel;
        if (k > T)      sel = true;
        else if (k == T){ sel = (rank < r); rank++; }
        else            sel = false;

        float v = 0.0f;
        if (sel) {
            float f = (k & 0x80000000u) ? __uint_as_float(k & 0x7FFFFFFFu)
                                        : __uint_as_float(~k);
            v = (f > 0.0f) ? f : 0.0f;                 // relu(topk)
            unsigned int slot = atomicAdd(&sh_nsel, 1u);
            pairs[(size_t)row * (2 * KSEL) + slot * 2    ] = __uint_as_float((unsigned int)idx);
            pairs[(size_t)row * (2 * KSEL) + slot * 2 + 1] = v;
        }
        hrow[idx] = v;   // h_sparse: relu(topk) at selected, 0 elsewhere
    }
}

// ---------------------------------------------------------------------------
// Kernel 3: sparse decode  x_hat = sum_j val_j * W_dec[idx_j, :] + b_dec
// ---------------------------------------------------------------------------
__global__ __launch_bounds__(256)
void sae_decode(const float* __restrict__ pairs,
                const float* __restrict__ Wdec,
                const float* __restrict__ bdec,
                float* __restrict__ xhat)
{
    __shared__ int   sidx[KSEL];
    __shared__ float sval[KSEL];
    const int row = blockIdx.x;
    const int tid = threadIdx.x;

    if (tid < KSEL) {
        sidx[tid] = (int)__float_as_uint(pairs[(size_t)row * (2 * KSEL) + tid * 2]);
        sval[tid] = pairs[(size_t)row * (2 * KSEL) + tid * 2 + 1];
    }
    __syncthreads();

    for (int c = tid; c < D_IN; c += 256) {
        float acc = bdec[c];
        #pragma unroll 8
        for (int j = 0; j < KSEL; ++j)
            acc += sval[j] * Wdec[(size_t)sidx[j] * D_IN + c];
        xhat[(size_t)row * D_IN + c] = acc;
    }
}

// ---------------------------------------------------------------------------
extern "C" void kernel_launch(void* const* d_in, const int* in_sizes, int n_in,
                              void* d_out, int out_size, void* d_ws, size_t ws_size,
                              hipStream_t stream)
{
    const float* x    = (const float*)d_in[0];
    const float* Wenc = (const float*)d_in[1];
    const float* benc = (const float*)d_in[2];
    const float* Wdec = (const float*)d_in[3];
    const float* bdec = (const float*)d_in[4];
    // d_in[5] is k (=64), baked in as KSEL.

    float* out   = (float*)d_out;
    float* xhat  = out;                               // [4096, 768]
    float* hsp   = out + (size_t)BATCH * D_IN;        // [4096, 24576]
    float* pairs = (float*)d_ws;                      // [4096, 64, 2] = 2 MB

    dim3 g1(D_SAE / BN, BATCH / BM);                  // 192 x 64
    sae_encode_gemm<<<g1, 256, 0, stream>>>(x, Wenc, benc, bdec, hsp);

    sae_topk_select<<<BATCH, 256, D_SAE * sizeof(unsigned int), stream>>>(hsp, pairs);

    sae_decode<<<BATCH, 256, 0, stream>>>(pairs, Wdec, bdec, xhat);
}
